// C2VecLayer_82867099009931
// MI455X (gfx1250) — matrix operand order repaired and checked by reading, not compiled, not run on hardware
//
#include <hip/hip_runtime.h>

// ---------------------------------------------------------------------------
// CDNA5 / gfx1250 bf16 WMMA bidirectional-LSTM.
//   D = 1024, T = 1024, B = 2 sequences, 2 directions.
//   Phase 1: zpre = X@W + b  (big parallel bf16 WMMA GEMM, f32 accum)
//   Phase 2: persistent kernel, 1024 steps, h@U via WMMA, one grid barrier
//            per step; cell state lives in LDS; h double-buffered in global
//            memory already laid out as WMMA A-fragments.
// ---------------------------------------------------------------------------

typedef __attribute__((ext_vector_type(16))) __bf16 v16bf;
typedef __attribute__((ext_vector_type(8)))  float  v8f;

#define SEQ   1024
#define DIM   1024
#define NWG_REC 128     // 2 dirs * 64 column-tiles

// workspace layout (bytes, all 256-aligned)
#define XBF_OFF    (size_t)0                       // 2*1024*1024 bf16 = 4 MB
#define WPACK_OFF  (size_t)(4u  * 1024 * 1024)     // 2 * 4M bf16      = 16 MB
#define UPACK_OFF  (size_t)(20u * 1024 * 1024)     // 2 * 4M bf16      = 16 MB
#define ZPRE_OFF   (size_t)(36u * 1024 * 1024)     // 2*2048*4096 f32  = 64 MB
#define HFRAG_OFF  (size_t)(100u * 1024 * 1024)    // 2buf*2dir*16384 bf16 = 128 KB
#define BAR_OFF    (size_t)(100u * 1024 * 1024 + 256u * 1024)

// ---------------------------------------------------------------------------
// init: zero the h fragment buffers (h0 = 0, and padded rows M>=2 stay 0
// forever) and the grid-barrier counter.  Runs every launch -> deterministic.
// ---------------------------------------------------------------------------
__global__ void init_ws(__bf16* __restrict__ hfrag, unsigned* __restrict__ bar) {
  int i = blockIdx.x * blockDim.x + threadIdx.x;   // 65536 threads
  if (i < 65536) hfrag[i] = (__bf16)0.0f;
  if (i == 0) bar[0] = 0u;
}

// ---------------------------------------------------------------------------
// x (f32) -> bf16, flat 2*1024*1024 elements, row r = s*1024+t, col d.
// ---------------------------------------------------------------------------
__global__ void convert_x(const float* __restrict__ x, __bf16* __restrict__ xbf) {
  size_t i = (size_t)blockIdx.x * blockDim.x + threadIdx.x;  // 2M
  xbf[i] = (__bf16)x[i];
}

// ---------------------------------------------------------------------------
// Pack W/U (1024 x 4096 f32, row-major) into WMMA B-fragment layout:
//   out[dir][ntile(256)][kb(32)][lane(32)][e(16)]  bf16
// B tile is K(32) x N(16): lane = 16*half + n, element e<8 -> K = 8*half + e, e>=8 -> K = 16 + 8*half + (e-8).
// mat: 0=Wf 1=Wb 2=Uf 3=Ub
// ---------------------------------------------------------------------------
__global__ void pack_weights(const float* __restrict__ Wf, const float* __restrict__ Wb,
                             const float* __restrict__ Uf, const float* __restrict__ Ub,
                             __bf16* __restrict__ wpack, __bf16* __restrict__ upack) {
  size_t o = (size_t)blockIdx.x * blockDim.x + threadIdx.x;  // 16,777,216
  int e    = (int)(o & 15);
  int lane = (int)((o >> 4) & 31);
  int kb   = (int)((o >> 9) & 31);
  int nt   = (int)((o >> 14) & 255);
  int mat  = (int)(o >> 22);
  int k = kb * 32 + 8 * (lane >> 4) + (e & 7) + ((e & 8) << 1);
  int n = nt * 16 + (lane & 15);
  const float* src = (mat == 0) ? Wf : (mat == 1) ? Wb : (mat == 2) ? Uf : Ub;
  __bf16* dst = (mat < 2) ? wpack : upack;
  size_t dbase = (size_t)(mat & 1) * 4194304u;
  dst[dbase + (o & 4194303u)] = (__bf16)src[(size_t)k * 4096 + n];
}

// ---------------------------------------------------------------------------
// zpre[dir][r][n] = (X @ W_dir)[r][n] + b_dir[n],  r in [0,2048), n in [0,4096)
// One wave per 16x16 output tile, K-loop of 32 bf16 WMMAs.
// A fragment (16x32 bf16): lane L -> row m0+(L&15); e<8 -> K=8*half+e,
// e>=8 -> K=16+8*half+(e-8): two contiguous 16-byte runs per lane.
// ---------------------------------------------------------------------------
__global__ __launch_bounds__(256) void zpre_gemm(
    const __bf16* __restrict__ xbf,     // [2048][1024]
    const __bf16* __restrict__ wpack,   // [2][256][32][512]
    const float*  __restrict__ bias_f,
    const float*  __restrict__ bias_b,
    float* __restrict__ zpre) {         // [2][2048][4096]
  int w    = blockIdx.x * 8 + (threadIdx.x >> 5);   // 65536 waves
  int lane = threadIdx.x & 31;
  int dir  = w >> 15;
  int rem  = w & 32767;
  int mt   = rem >> 8;      // 0..127
  int nt   = rem & 255;     // 0..255
  int half = lane >> 4;
  int rr   = mt * 16 + (lane & 15);

  v8f acc = {};
  const __bf16* wb   = wpack + ((size_t)dir * 256 + nt) * (32 * 512);
  const __bf16* xrow = xbf + (size_t)rr * 1024;
  for (int kb = 0; kb < 32; ++kb) {
    union { v16bf v; uint4 q[2]; } a;
    a.q[0] = *(const uint4*)(xrow + kb * 32 + 8 * half);
    a.q[1] = *(const uint4*)(xrow + kb * 32 + 16 + 8 * half);
    v16bf b = *(const v16bf*)(wb + kb * 512 + lane * 16);
    acc = __builtin_amdgcn_wmma_f32_16x16x32_bf16(false, a.v, false, b,
                                                  (short)0, acc, false, false);
  }
  int n = nt * 16 + (lane & 15);
  float bv = (dir ? bias_b : bias_f)[n];
#pragma unroll
  for (int v = 0; v < 8; ++v) {
    int r = mt * 16 + v + 8 * half;   // C layout: lane element v -> M = v+8*half
    zpre[((size_t)dir * 2048 + r) * 4096 + n] = acc[v] + bv;
  }
}

// ---------------------------------------------------------------------------
// Persistent recurrent kernel: 128 WGs x 128 threads (4 waves).
// WG (dir, jt) owns hidden columns [jt*16, jt*16+16) of direction dir.
// Wave wv handles K range [wv*256, wv*256+256) for all 4 gates; partials
// reduced through LDS; threads 0..31 do the pointwise LSTM update.
// One monotonic-counter grid barrier per step.
// ---------------------------------------------------------------------------
__global__ __launch_bounds__(128) void lstm_recurrent(
    const __bf16* __restrict__ upack,   // [2][256][32][512]
    const float*  __restrict__ zpre,    // [2][2048][4096]
    __bf16* __restrict__ hfrag,         // [2buf][2dir][32kb][512]
    float* __restrict__ out,            // [2][1024][2048]
    unsigned* __restrict__ bar) {
  __shared__ float part[4][4][2][16];   // [gate][wave][row][col]
  __shared__ float cstate[2][16];

  const int tid  = threadIdx.x;
  const int lane = tid & 31;
  const int wv   = tid >> 5;
  const int dir  = blockIdx.x >> 6;
  const int jt   = blockIdx.x & 63;

  if (tid < 32) cstate[tid >> 4][tid & 15] = 0.0f;
  __syncthreads();

  const __bf16* ubase = upack + (size_t)dir * (256 * 32 * 512);
  const int laneoff = lane * 16;

  for (int t = 0; t < SEQ; ++t) {
    const int buf_r = t & 1;
    v8f zero = {};
    v8f acc[4];
#pragma unroll
    for (int g = 0; g < 4; ++g) acc[g] = zero;

    const __bf16* hbase = hfrag + (size_t)(buf_r * 2 + dir) * (32 * 512);
#pragma unroll
    for (int i = 0; i < 8; ++i) {
      int kb = wv * 8 + i;
      v16bf a = *(const v16bf*)(hbase + kb * 512 + laneoff);  // A frag (pre-swizzled)
#pragma unroll
      for (int g = 0; g < 4; ++g) {
        int nt = g * 64 + jt;                                 // gate g, cols jt*16..
        v16bf b = *(const v16bf*)(ubase + ((size_t)nt * 32 + kb) * 512 + laneoff);
        acc[g] = __builtin_amdgcn_wmma_f32_16x16x32_bf16(false, a, false, b,
                                                         (short)0, acc[g], false, false);
      }
    }

    // C layout: row0 = element 0 of lanes 0..15, row1 = element 1.
    if (lane < 16) {
#pragma unroll
      for (int g = 0; g < 4; ++g) {
        part[g][wv][0][lane] = acc[g][0];
        part[g][wv][1][lane] = acc[g][1];
      }
    }
    __syncthreads();

    if (tid < 32) {
      int m = tid >> 4, col = tid & 15;
      int tt = dir ? (SEQ - 1 - t) : t;                 // backward dir runs reversed
      size_t zrow = ((size_t)dir * 2048 + m * 1024 + tt) * 4096 + jt * 16 + col;
      float gate[4];
#pragma unroll
      for (int g = 0; g < 4; ++g) {
        float s = zpre[zrow + g * 1024]
                + part[g][0][m][col] + part[g][1][m][col]
                + part[g][2][m][col] + part[g][3][m][col];
        gate[g] = 1.0f / (1.0f + __expf(-s));           // all activations sigmoid
      }
      float c = gate[1] * cstate[m][col] + gate[0] * gate[2];
      cstate[m][col] = c;
      float h = gate[3] * (1.0f / (1.0f + __expf(-c)));

      out[((size_t)m * SEQ + tt) * 2048 + dir * 1024 + jt * 16 + col] = h;

      // write h into next step's A-fragment buffer (bf16, swizzled layout):
      // d -> kb = d>>5, K = d&31, half = (K>>3)&1, e = (K&7)|((K&16)>>1),
      // lane = 16*half + m.
      int d  = jt * 16 + col;
      int kb = d >> 5, K = d & 31;
      int hf = (K >> 3) & 1;
      int e  = (K & 7) | ((K & 16) >> 1);
      int wl = hf * 16 + m;
      hfrag[((size_t)(((t + 1) & 1) * 2 + dir) * 32 + kb) * 512 + wl * 16 + e] = (__bf16)h;
    }

    // ---- grid barrier (monotonic counter; re-zeroed each launch) ----
    __syncthreads();
    if (tid == 0) {
      __threadfence();
      __hip_atomic_fetch_add(bar, 1u, __ATOMIC_ACQ_REL, __HIP_MEMORY_SCOPE_AGENT);
      unsigned target = (unsigned)(t + 1) * NWG_REC;
      while (__hip_atomic_load(bar, __ATOMIC_ACQUIRE, __HIP_MEMORY_SCOPE_AGENT) < target) {
        __builtin_amdgcn_s_sleep(1);
      }
      __threadfence();
    }
    __syncthreads();
  }
}

// ---------------------------------------------------------------------------
extern "C" void kernel_launch(void* const* d_in, const int* in_sizes, int n_in,
                              void* d_out, int out_size, void* d_ws, size_t ws_size,
                              hipStream_t stream) {
  const float* x  = (const float*)d_in[0];
  const float* Wf = (const float*)d_in[1];
  const float* Uf = (const float*)d_in[2];
  const float* bf = (const float*)d_in[3];
  const float* Wb = (const float*)d_in[4];
  const float* Ub = (const float*)d_in[5];
  const float* bb = (const float*)d_in[6];
  float* out = (float*)d_out;

  char* ws = (char*)d_ws;
  __bf16*   xbf   = (__bf16*)(ws + XBF_OFF);
  __bf16*   wpack = (__bf16*)(ws + WPACK_OFF);
  __bf16*   upack = (__bf16*)(ws + UPACK_OFF);
  float*    zpre  = (float*)(ws + ZPRE_OFF);
  __bf16*   hfrag = (__bf16*)(ws + HFRAG_OFF);
  unsigned* bar   = (unsigned*)(ws + BAR_OFF);

  init_ws<<<256, 256, 0, stream>>>(hfrag, bar);
  convert_x<<<8192, 256, 0, stream>>>(x, xbf);
  pack_weights<<<65536, 256, 0, stream>>>(Wf, Wb, Uf, Ub, wpack, upack);
  zpre_gemm<<<8192, 256, 0, stream>>>(xbf, wpack, bf, bb, zpre);
  lstm_recurrent<<<NWG_REC, 128, 0, stream>>>(upack, zpre, hfrag, out, bar);
}
